// SegformerGAT_90460601189006
// MI455X (gfx1250) — matrix operand order repaired and checked by reading, not compiled, not run on hardware
//
#include <hip/hip_runtime.h>
#include <hip/hip_bf16.h>

// ---------------------------------------------------------------------------
// Problem constants (match reference)
// ---------------------------------------------------------------------------
#define BATCH   2
#define GRID_D  128
#define NNODE   (GRID_D * GRID_D)      // 16384
#define NROWS   (BATCH * NNODE)        // 32768
#define C_IN    512
#define HID     128
#define HEADS   4
#define C_OUT   128
#define LRELU   0.2f
#define EPS_LN  1e-5f

typedef __attribute__((ext_vector_type(16))) _Float16 v16h;
typedef __attribute__((ext_vector_type(8)))  _Float16 v8h;
typedef __attribute__((ext_vector_type(8)))  float    v8f;

// ---------------------------------------------------------------------------
// CDNA5 async-to-LDS helpers (ASYNCcnt path, CDNA5 ISA).
// GVS mode: SGPR base + VGPR offset address global memory, a VGPR holds the LDS address; imm offset applies to both
// ---------------------------------------------------------------------------
__device__ __forceinline__ void async_copy_32B(unsigned lds_addr, unsigned goff,
                                               const void* base) {
    asm volatile(
        "global_load_async_to_lds_b128 %0, %1, %2\n\t"
        "global_load_async_to_lds_b128 %0, %1, %2 offset:16"
        :
        : "v"(lds_addr), "v"(goff), "s"(base)
        : "memory");
}
__device__ __forceinline__ void wait_async0() {
    asm volatile("s_wait_asynccnt 0x0" ::: "memory");
}

// ---------------------------------------------------------------------------
// Weight packing: f32 [K][N] (or transposed src [N][K]) -> f16 WMMA B-fragment
// order.  Layout: tile (kt,nt) -> 32 lanes -> 16 contiguous f16 per lane.
// ---------------------------------------------------------------------------
__global__ __launch_bounds__(256)
void pack_b_kernel(const float* __restrict__ src, _Float16* __restrict__ dst,
                   int K, int N, int transposed_src) {
    int gid = blockIdx.x * blockDim.x + threadIdx.x;
    if (gid >= K * N) return;
    int i    = gid & 15;
    int lane = (gid >> 4) & 31;
    int tile = gid >> 9;
    int nt_n = N >> 4;
    int kt   = tile / nt_n;
    int nt   = tile - kt * nt_n;
    int n = nt * 16 + (lane & 15);
    int k = kt * 32 + ((lane >> 4) << 3) + (i & 7) + ((i >> 3) << 4);
    float v = transposed_src ? src[(size_t)n * K + k] : src[(size_t)k * N + n];
    dst[gid] = (_Float16)v;
}

// ---------------------------------------------------------------------------
// A-fragment loaders (ISA 16-bit A 16x32: lanes<16 K{0..7,16..23},
// lanes>=16 K{8..15,24..31}).
// ---------------------------------------------------------------------------
__device__ __forceinline__ v16h load_a_frag(const float* __restrict__ A,
                                            int ld, int r0, int kbase, int lane) {
    int row = r0 + (lane & 15);
    int kk  = kbase + ((lane >> 4) << 3);
    const float* p = A + (size_t)row * ld + kk;
    v16h f;
#pragma unroll
    for (int i = 0; i < 8; ++i) f[i]     = (_Float16)p[i];
#pragma unroll
    for (int i = 0; i < 8; ++i) f[8 + i] = (_Float16)p[16 + i];
    return f;
}
__device__ __forceinline__ v16h load_a_frag(const _Float16* __restrict__ A,
                                            int ld, int r0, int kbase, int lane) {
    int row = r0 + (lane & 15);
    int kk  = kbase + ((lane >> 4) << 3);
    const _Float16* p = A + (size_t)row * ld + kk;
    v8h lo = *(const v8h*)p;
    v8h hi = *(const v8h*)(p + 16);
    v16h f;
#pragma unroll
    for (int i = 0; i < 8; ++i) { f[i] = lo[i]; f[8 + i] = hi[i]; }
    return f;
}

// ---------------------------------------------------------------------------
// GEMM + bias + LayerNorm + ReLU, f16 output (consumed as next GEMM's A).
// ---------------------------------------------------------------------------
template <int K, int NCOL, bool CONCAT, typename AT>
__global__ __launch_bounds__(256)
void gemm_ln_kernel(const AT* __restrict__ A0, const AT* __restrict__ A1,
                    const _Float16* __restrict__ Bp,
                    const float* __restrict__ bias,
                    const float* __restrict__ gamma, const float* __restrict__ beta,
                    _Float16* __restrict__ out) {
    constexpr int NT  = NCOL / 16;
    constexpr int KS  = K / 32;
    constexpr int TPW = NT / 8;
    static_assert(NT % 8 == 0, "NT must be a multiple of 8");

    const int tid  = threadIdx.x;
    const int wave = tid >> 5;
    const int lane = tid & 31;
    const int r0   = blockIdx.x * 16;

    const v16h* __restrict__ Bfrag = (const v16h*)Bp;

    v8f acc[TPW] = {};
    for (int ks = 0; ks < KS; ++ks) {
        const int kb = ks * 32;
        v16h af;
        if constexpr (CONCAT) {
            if (kb < K / 2) af = load_a_frag(A0, K / 2, r0, kb, lane);
            else            af = load_a_frag(A1, K / 2, r0, kb - K / 2, lane);
        } else {
            af = load_a_frag(A0, K, r0, kb, lane);
        }
        v16h bf[TPW];
#pragma unroll
        for (int t = 0; t < TPW; ++t)
            bf[t] = Bfrag[(size_t)(ks * NT + wave + t * 8) * 32 + lane];
#pragma unroll
        for (int t = 0; t < TPW; ++t)
            acc[t] = __builtin_amdgcn_wmma_f32_16x16x32_f16(
                false, af, false, bf[t], (short)0, acc[t], false, false);
    }

    // epilogue through LDS: bias add, per-row LayerNorm + ReLU, f16 store
    __shared__ float smem[16][NCOL + 4];
#pragma unroll
    for (int t = 0; t < TPW; ++t) {
        const int col   = (wave + t * 8) * 16 + (lane & 15);
        const int rbase = (lane >> 4) << 3;
        const float bv  = bias[col];
#pragma unroll
        for (int r = 0; r < 8; ++r) smem[rbase + r][col] = acc[t][r] + bv;
    }
    __syncthreads();

    __shared__ float red[16][16], red2[16][16];
    {
        const int row = tid >> 4, sl = tid & 15;
        float s = 0.f, s2 = 0.f;
#pragma unroll
        for (int j = 0; j < NCOL / 16; ++j) {
            float v = smem[row][sl + j * 16];
            s += v; s2 += v * v;
        }
        red[row][sl] = s; red2[row][sl] = s2;
    }
    __syncthreads();
    __shared__ float mu_s[16], rs_s[16];
    if (tid < 16) {
        float s = 0.f, s2 = 0.f;
#pragma unroll
        for (int j = 0; j < 16; ++j) { s += red[tid][j]; s2 += red2[tid][j]; }
        const float m   = s * (1.f / NCOL);
        const float var = s2 * (1.f / NCOL) - m * m;
        mu_s[tid] = m;
        rs_s[tid] = rsqrtf(var + EPS_LN);
    }
    __syncthreads();
    for (int idx = tid; idx < 16 * NCOL; idx += 256) {
        const int row = idx / NCOL, col = idx % NCOL;
        float v = (smem[row][col] - mu_s[row]) * rs_s[row] * gamma[col] + beta[col];
        out[(size_t)(r0 + row) * NCOL + col] = (_Float16)fmaxf(v, 0.f);
    }
}

// ---------------------------------------------------------------------------
// Dual GEMM (shared f16 A, two packed Bs) + bias -> two f32 outputs.
// The shared 16-row A tile is staged once per block into LDS with
// global_load_async_to_lds_b128 (double buffered, s_wait_asynccnt + barrier),
// then every wave rebuilds its fragment with ds_load_b128.
// ---------------------------------------------------------------------------
template <int K, int NCOL>
__global__ __launch_bounds__(256)
void gemm_dual_kernel(const _Float16* __restrict__ A,
                      const _Float16* __restrict__ Bp0, const _Float16* __restrict__ Bp1,
                      const float* __restrict__ bias0, const float* __restrict__ bias1,
                      float* __restrict__ out0, float* __restrict__ out1) {
    constexpr int NT  = NCOL / 16;
    constexpr int TOT = 2 * NT;
    constexpr int TPW = TOT / 8;
    constexpr int KS  = K / 32;
    static_assert(TOT % 8 == 0, "tile count must be a multiple of 8");

    const int tid  = threadIdx.x;
    const int wave = tid >> 5;
    const int lane = tid & 31;
    const int r0   = blockIdx.x * 16;

    __shared__ __align__(16) _Float16 aTile[2][16 * 32];   // 2 x 1KB
    const unsigned aBase = (unsigned)(uintptr_t)(&aTile[0][0]);

    const int srow  = lane & 15;          // staging: row
    const int shalf = lane >> 4;          // staging: 16-element half
    auto stage = [&](int ks, int b) {     // wave 0 only
        unsigned ldsa = aBase + (unsigned)(b * (16 * 32) + srow * 32 + shalf * 16) * 2u;
        unsigned goff = (unsigned)((r0 + srow) * K + ks * 32 + shalf * 16) * 2u;
        async_copy_32B(ldsa, goff, A);
    };

    if (wave == 0) { stage(0, 0); wait_async0(); }
    __syncthreads();

    v8f acc[TPW] = {};
    for (int ks = 0; ks < KS; ++ks) {
        if (wave == 0 && ks + 1 < KS) stage(ks + 1, (ks + 1) & 1);

        // rebuild A fragment from LDS (ds_load_b128 x2)
        const _Float16* ap = &aTile[ks & 1][(lane & 15) * 32 + ((lane >> 4) << 3)];
        v8h lo = *(const v8h*)ap;
        v8h hi = *(const v8h*)(ap + 16);
        v16h af;
#pragma unroll
        for (int i = 0; i < 8; ++i) { af[i] = lo[i]; af[8 + i] = hi[i]; }

        v16h bf[TPW];
#pragma unroll
        for (int t = 0; t < TPW; ++t) {
            const int gt    = wave + t * 8;
            const int which = (gt >= NT) ? 1 : 0;
            const int nt    = which ? gt - NT : gt;
            const v16h* __restrict__ Bf = (const v16h*)(which ? Bp1 : Bp0);
            bf[t] = Bf[(size_t)(ks * NT + nt) * 32 + lane];
        }
#pragma unroll
        for (int t = 0; t < TPW; ++t)
            acc[t] = __builtin_amdgcn_wmma_f32_16x16x32_f16(
                false, af, false, bf[t], (short)0, acc[t], false, false);

        if (wave == 0 && ks + 1 < KS) wait_async0();
        __syncthreads();
    }
#pragma unroll
    for (int t = 0; t < TPW; ++t) {
        const int gt    = wave + t * 8;
        const int which = (gt >= NT) ? 1 : 0;
        const int nt    = which ? gt - NT : gt;
        const int col   = nt * 16 + (lane & 15);
        const int rbase = (lane >> 4) << 3;
        const float bv  = (which ? bias1 : bias0)[col];
        float* __restrict__ o = which ? out1 : out0;
#pragma unroll
        for (int r = 0; r < 8; ++r)
            o[(size_t)(r0 + rbase + r) * NCOL + col] = acc[t][r] + bv;
    }
}

// ---------------------------------------------------------------------------
// Final classifier: out[b][o][n] = g2 @ Wc^T + bc, transposed NCHW store
// (WMMA D layout: lane = column o, consecutive VGPRs = consecutive rows n).
// ---------------------------------------------------------------------------
__global__ __launch_bounds__(256)
void final_gemm_kernel(const _Float16* __restrict__ A, const _Float16* __restrict__ Bp,
                       const float* __restrict__ bc, float* __restrict__ out) {
    constexpr int K = C_OUT, NCOL = C_IN, NT = NCOL / 16, TPW = NT / 8, KS = K / 32;
    const int tid  = threadIdx.x;
    const int wave = tid >> 5;
    const int lane = tid & 31;
    const int r0   = blockIdx.x * 16;

    const v16h* __restrict__ Bfrag = (const v16h*)Bp;

    v16h af[KS];
#pragma unroll
    for (int ks = 0; ks < KS; ++ks) af[ks] = load_a_frag(A, K, r0, ks * 32, lane);

    v8f acc[TPW] = {};
#pragma unroll
    for (int t = 0; t < TPW; ++t) {
        const int nt = wave + t * 8;
        v16h bf[KS];
#pragma unroll
        for (int ks = 0; ks < KS; ++ks)
            bf[ks] = Bfrag[(size_t)(ks * NT + nt) * 32 + lane];
#pragma unroll
        for (int ks = 0; ks < KS; ++ks)
            acc[t] = __builtin_amdgcn_wmma_f32_16x16x32_f16(
                false, af[ks], false, bf[ks], (short)0, acc[t], false, false);
    }

    const int b  = r0 >> 14;
    const int n0 = (r0 & (NNODE - 1)) + ((lane >> 4) << 3);
#pragma unroll
    for (int t = 0; t < TPW; ++t) {
        const int o  = (wave + t * 8) * 16 + (lane & 15);
        const float bv = bc[o];
        float* dst = out + (((size_t)b * C_IN + o) << 14) + n0;
        float4 v0 = {acc[t][0] + bv, acc[t][1] + bv, acc[t][2] + bv, acc[t][3] + bv};
        float4 v1 = {acc[t][4] + bv, acc[t][5] + bv, acc[t][6] + bv, acc[t][7] + bv};
        ((float4*)dst)[0] = v0;
        ((float4*)dst)[1] = v1;
    }
}

// ---------------------------------------------------------------------------
// GATv2 edge softmax, layer 1 (4 heads x 128 ch), f16 output.
// ---------------------------------------------------------------------------
__global__ __launch_bounds__(128)
void gat_edge4_kernel(const float* __restrict__ xl, const float* __restrict__ xr,
                      const float* __restrict__ att, const float* __restrict__ bias,
                      _Float16* __restrict__ out) {
    const int node = blockIdx.x;
    const int t    = threadIdx.x;
    const int bI   = node >> 14;
    const int nl   = node & (NNODE - 1);
    const int gy   = nl >> 7, gx = nl & (GRID_D - 1);

    float xr4[HEADS], at4[HEADS];
#pragma unroll
    for (int h = 0; h < HEADS; ++h) {
        xr4[h] = xr[(size_t)node * (HEADS * HID) + h * HID + t];
        at4[h] = att[h * HID + t];
    }

    __shared__ float logit[9][HEADS];
    __shared__ float wred[4][HEADS];

    const int offs[9][2] = {{-1,-1},{-1,0},{-1,1},{0,-1},{0,0},{0,1},{1,-1},{1,0},{1,1}};

    for (int e = 0; e < 9; ++e) {
        const int yy = gy + offs[e][0], xx = gx + offs[e][1];
        const bool valid = (yy >= 0) & (yy < GRID_D) & (xx >= 0) & (xx < GRID_D);
        const int s = valid ? ((bI << 14) + (yy << 7) + xx) : node;
        float p[HEADS];
#pragma unroll
        for (int h = 0; h < HEADS; ++h) {
            float v = xl[(size_t)s * (HEADS * HID) + h * HID + t] + xr4[h];
            v = (v > 0.f) ? v : LRELU * v;
            p[h] = v * at4[h];
        }
#pragma unroll
        for (int h = 0; h < HEADS; ++h)
#pragma unroll
            for (int off = 16; off > 0; off >>= 1)
                p[h] += __shfl_xor(p[h], off, 32);
        if ((t & 31) == 0) {
            const int w = t >> 5;
#pragma unroll
            for (int h = 0; h < HEADS; ++h) wred[w][h] = p[h];
        }
        __syncthreads();
        if (t < HEADS)
            logit[e][t] = valid ? (wred[0][t] + wred[1][t] + wred[2][t] + wred[3][t])
                                : -1e30f;
        __syncthreads();
    }

    float mx[HEADS], den[HEADS], accv[HEADS];
#pragma unroll
    for (int h = 0; h < HEADS; ++h) { mx[h] = -1e30f; den[h] = 0.f; accv[h] = 0.f; }
    for (int e = 0; e < 9; ++e)
#pragma unroll
        for (int h = 0; h < HEADS; ++h) mx[h] = fmaxf(mx[h], logit[e][h]);
    for (int e = 0; e < 9; ++e)
#pragma unroll
        for (int h = 0; h < HEADS; ++h) den[h] += __expf(logit[e][h] - mx[h]);

    for (int e = 0; e < 9; ++e) {
        const int yy = gy + offs[e][0], xx = gx + offs[e][1];
        const bool valid = (yy >= 0) & (yy < GRID_D) & (xx >= 0) & (xx < GRID_D);
        const int s = valid ? ((bI << 14) + (yy << 7) + xx) : node;
#pragma unroll
        for (int h = 0; h < HEADS; ++h) {
            const float alpha = __expf(logit[e][h] - mx[h]) / den[h];   // 0 if invalid
            accv[h] += alpha * xl[(size_t)s * (HEADS * HID) + h * HID + t];
        }
    }
#pragma unroll
    for (int h = 0; h < HEADS; ++h) {
        float v = accv[h] + bias[h * HID + t];
        out[(size_t)node * (HEADS * HID) + h * HID + t] = (_Float16)fmaxf(v, 0.f);
    }
}

// ---------------------------------------------------------------------------
// GATv2 edge softmax, layer 2 (1 head x 128 ch), f16 output.
// ---------------------------------------------------------------------------
__global__ __launch_bounds__(128)
void gat_edge1_kernel(const float* __restrict__ xl, const float* __restrict__ xr,
                      const float* __restrict__ att, const float* __restrict__ bias,
                      _Float16* __restrict__ out) {
    const int node = blockIdx.x;
    const int t    = threadIdx.x;
    const int bI   = node >> 14;
    const int nl   = node & (NNODE - 1);
    const int gy   = nl >> 7, gx = nl & (GRID_D - 1);

    const float xrv = xr[(size_t)node * C_OUT + t];
    const float atv = att[t];

    __shared__ float logit[9];
    __shared__ float wred[4];

    const int offs[9][2] = {{-1,-1},{-1,0},{-1,1},{0,-1},{0,0},{0,1},{1,-1},{1,0},{1,1}};

    for (int e = 0; e < 9; ++e) {
        const int yy = gy + offs[e][0], xx = gx + offs[e][1];
        const bool valid = (yy >= 0) & (yy < GRID_D) & (xx >= 0) & (xx < GRID_D);
        const int s = valid ? ((bI << 14) + (yy << 7) + xx) : node;
        float v = xl[(size_t)s * C_OUT + t] + xrv;
        v = (v > 0.f) ? v : LRELU * v;
        float p = v * atv;
#pragma unroll
        for (int off = 16; off > 0; off >>= 1) p += __shfl_xor(p, off, 32);
        if ((t & 31) == 0) wred[t >> 5] = p;
        __syncthreads();
        if (t == 0) logit[e] = valid ? (wred[0] + wred[1] + wred[2] + wred[3]) : -1e30f;
        __syncthreads();
    }

    float mx = -1e30f, den = 0.f, accv = 0.f;
    for (int e = 0; e < 9; ++e) mx = fmaxf(mx, logit[e]);
    for (int e = 0; e < 9; ++e) den += __expf(logit[e] - mx);
    for (int e = 0; e < 9; ++e) {
        const int yy = gy + offs[e][0], xx = gx + offs[e][1];
        const bool valid = (yy >= 0) & (yy < GRID_D) & (xx >= 0) & (xx < GRID_D);
        const int s = valid ? ((bI << 14) + (yy << 7) + xx) : node;
        const float alpha = __expf(logit[e] - mx) / den;
        accv += alpha * xl[(size_t)s * C_OUT + t];
    }
    float v = accv + bias[t];
    out[(size_t)node * C_OUT + t] = (_Float16)fmaxf(v, 0.f);
}

// ---------------------------------------------------------------------------
// Host launcher
// ---------------------------------------------------------------------------
extern "C" void kernel_launch(void* const* d_in, const int* in_sizes, int n_in,
                              void* d_out, int out_size, void* d_ws, size_t ws_size,
                              hipStream_t stream) {
    const float* rgb   = (const float*)d_in[0];
    const float* xfeat = (const float*)d_in[1];
    const float* Wf    = (const float*)d_in[2];
    const float* bf    = (const float*)d_in[3];
    const float* gf    = (const float*)d_in[4];
    const float* betaf = (const float*)d_in[5];
    const float* W_in  = (const float*)d_in[6];
    const float* b_in  = (const float*)d_in[7];
    const float* gn    = (const float*)d_in[8];
    const float* betan = (const float*)d_in[9];
    const float* Wl1   = (const float*)d_in[10];
    const float* bl1   = (const float*)d_in[11];
    const float* Wr1   = (const float*)d_in[12];
    const float* br1   = (const float*)d_in[13];
    const float* att1  = (const float*)d_in[14];
    const float* bias1 = (const float*)d_in[15];
    const float* Wl2   = (const float*)d_in[16];
    const float* bl2   = (const float*)d_in[17];
    const float* Wr2   = (const float*)d_in[18];
    const float* br2   = (const float*)d_in[19];
    const float* att2  = (const float*)d_in[20];
    const float* bias2 = (const float*)d_in[21];
    const float* Wc    = (const float*)d_in[22];
    const float* bc    = (const float*)d_in[23];
    float* outp        = (float*)d_out;

    char* ws = (char*)d_ws;
    size_t off = 0;
    auto carve = [&](size_t bytes) -> char* {
        char* p = ws + off;
        off += (bytes + 255) & ~(size_t)255;
        return p;
    };
    _Float16* pWf  = (_Float16*)carve((size_t)1024 * 512 * 2);
    _Float16* pWin = (_Float16*)carve((size_t)512 * 128 * 2);
    _Float16* pWl1 = (_Float16*)carve((size_t)128 * 512 * 2);
    _Float16* pWr1 = (_Float16*)carve((size_t)128 * 512 * 2);
    _Float16* pWl2 = (_Float16*)carve((size_t)512 * 128 * 2);
    _Float16* pWr2 = (_Float16*)carve((size_t)512 * 128 * 2);
    _Float16* pWc  = (_Float16*)carve((size_t)128 * 512 * 2);
    char* bufA = carve((size_t)NROWS * 512 * 2);   // fused (f16), later g1 (f16)
    char* bufB = carve((size_t)NROWS * 128 * 4);   // h (f16), later xl2 (f32)
    char* bufC = carve((size_t)NROWS * 512 * 4);   // xl1 (f32), later g2 (f16)
    char* bufD = carve((size_t)NROWS * 512 * 4);   // xr1 (f32)
    char* bufE = carve((size_t)NROWS * 128 * 4);   // xr2 (f32)

    // 1) pack weights into WMMA B-fragment order (f16)
    pack_b_kernel<<<(1024 * 512) / 256, 256, 0, stream>>>(Wf,  pWf,  1024, 512, 0);
    pack_b_kernel<<<(512 * 128) / 256, 256, 0, stream>>>(W_in, pWin, 512, 128, 0);
    pack_b_kernel<<<(128 * 512) / 256, 256, 0, stream>>>(Wl1,  pWl1, 128, 512, 0);
    pack_b_kernel<<<(128 * 512) / 256, 256, 0, stream>>>(Wr1,  pWr1, 128, 512, 0);
    pack_b_kernel<<<(512 * 128) / 256, 256, 0, stream>>>(Wl2,  pWl2, 512, 128, 0);
    pack_b_kernel<<<(512 * 128) / 256, 256, 0, stream>>>(Wr2,  pWr2, 512, 128, 0);
    pack_b_kernel<<<(128 * 512) / 256, 256, 0, stream>>>(Wc,   pWc,  128, 512, 1);

    const int RB = NROWS / 16;   // 2048 row blocks

    // 2) fused = relu(LN(concat(rgb,x) @ Wf + bf))          [f16 out]
    gemm_ln_kernel<1024, 512, true, float><<<RB, 256, 0, stream>>>(
        rgb, xfeat, pWf, bf, gf, betaf, (_Float16*)bufA);
    // 3) h = relu(LN(fused @ W_in + b_in))                  [f16 out]
    gemm_ln_kernel<512, 128, false, _Float16><<<RB, 256, 0, stream>>>(
        (const _Float16*)bufA, nullptr, pWin, b_in, gn, betan, (_Float16*)bufB);
    // 4) xl1 = h@Wl1+bl1 ; xr1 = h@Wr1+br1                  [f32 out]
    gemm_dual_kernel<128, 512><<<RB, 256, 0, stream>>>(
        (const _Float16*)bufB, pWl1, pWr1, bl1, br1, (float*)bufC, (float*)bufD);
    // 5) g1 = relu(GATv2 edge softmax, 4 heads)             [f16 out]
    gat_edge4_kernel<<<NROWS, 128, 0, stream>>>(
        (const float*)bufC, (const float*)bufD, att1, bias1, (_Float16*)bufA);
    // 6) xl2 = g1@Wl2+bl2 ; xr2 = g1@Wr2+br2                [f32 out]
    gemm_dual_kernel<512, 128><<<RB, 256, 0, stream>>>(
        (const _Float16*)bufA, pWl2, pWr2, bl2, br2, (float*)bufB, (float*)bufE);
    // 7) g2 = relu(GATv2 edge softmax, 1 head)              [f16 out]
    gat_edge1_kernel<<<NROWS, 128, 0, stream>>>(
        (const float*)bufB, (const float*)bufE, att2, bias2, (_Float16*)bufC);
    // 8) out[b][o][h][w] = g2 @ Wc^T + bc  (transposed store)
    final_gemm_kernel<<<RB, 256, 0, stream>>>((const _Float16*)bufC, pWc, bc, outp);
}
